// TailLayer_9929964389233
// MI455X (gfx1250) — compile-verified, matrix operand order checked by reading, not run on hardware
//
#include <hip/hip_runtime.h>

// ---------------------------------------------------------------------------
// MoE transformer tail layer for MI455X (gfx1250, wave32, WMMA, async-to-LDS).
// Compute-bound (~0.5 TFLOP vs ~0.5 GB traffic at 23.3 TB/s): all heavy math
// on v_wmma_f32_16x16x32_bf16 with f32 accumulation. Weights/activations are
// bf16-resident (one k_cvt pass / bf16 GEMM epilogues); GEMM tiles are staged
// with double-buffered global_load_async_to_lds_b128 (ASYNCcnt) so the async
// engine overlaps the WMMA bursts. Workspace requirement: ~460 MB.
// ---------------------------------------------------------------------------

#define B_  16
#define S_  512
#define D_  1024
#define H_  16
#define DH_ 64
#define DFF_ 4096
#define E_  3
#define U_  2
#define NI_ 2
#define TOK_ (B_ * S_)

typedef __attribute__((ext_vector_type(16))) __bf16 v16bf;
typedef __attribute__((ext_vector_type(8)))  __bf16 v8bf;
typedef __attribute__((ext_vector_type(4)))  __bf16 v4bf;
typedef __attribute__((ext_vector_type(2)))  __bf16 v2bf;
typedef __attribute__((ext_vector_type(8)))  float  v8f;

// ---------------------------------------------------------------------------
// CDNA5 helpers
// ---------------------------------------------------------------------------

// Async global->LDS copy of 16 bytes (raw, no VGPR data path, ASYNCcnt).
__device__ __forceinline__ void async_b128(const __bf16* gptr, __bf16* lptr) {
  unsigned lds = (unsigned)(size_t)lptr;           // low 32 bits = LDS offset
  unsigned long long ga = (unsigned long long)gptr;
  asm volatile("global_load_async_to_lds_b128 %0, %1, off"
               :: "v"(lds), "v"(ga) : "memory");
}

template <int N>
__device__ __forceinline__ void wait_async() {
#if __has_builtin(__builtin_amdgcn_s_wait_asynccnt)
  __builtin_amdgcn_s_wait_asynccnt(N);
#else
  asm volatile("s_wait_asynccnt %0" :: "i"(N) : "memory");
#endif
}

// Load a 16x32 bf16 fragment (A-layout) from LDS.
// ISA layout: lanes 0-15 hold row=lane, K 0-7 / 16-23; lanes 16-31 hold
// row=lane-16, K 8-15 / 24-31. Also serves as the B fragment when the LDS tile
// is stored transposed ("rows" = B columns, K contiguous).
__device__ __forceinline__ v16bf ld_frag(const __bf16* base, int stride) {
  const int lane = threadIdx.x & 31;
  const __bf16* p = base + (lane & 15) * stride + ((lane >> 4) << 3);
  v8bf lo = *(const v8bf*)(p);
  v8bf hi = *(const v8bf*)(p + 16);
  v16bf r;
#pragma unroll
  for (int i = 0; i < 8; ++i) { r[i] = lo[i]; r[i + 8] = hi[i]; }
  return r;
}

__device__ __forceinline__ v8f wmma_bf16(v16bf a, v16bf b, v8f c) {
  return __builtin_amdgcn_wmma_f32_16x16x32_bf16(
      false, a, false, b, (short)0, c, false, false);
}

__device__ __forceinline__ float gelu_tanh(float x) {
  const float k0 = 0.7978845608028654f, k1 = 0.044715f;
  return 0.5f * x * (1.0f + tanhf(k0 * (x + k1 * x * x * x)));
}

// ---------------------------------------------------------------------------
// Bulk f32 -> bf16 conversion (n must be a multiple of 8; 8 elems/thread).
// ---------------------------------------------------------------------------
__global__ __launch_bounds__(256) void k_cvt(const float* __restrict__ src,
                                             __bf16* __restrict__ dst,
                                             size_t n) {
  size_t i = ((size_t)blockIdx.x * 256 + threadIdx.x) * 8;
  if (i >= n) return;
  float4 a = *(const float4*)(src + i);
  float4 b = *(const float4*)(src + i + 4);
  v8bf o;
  o[0] = (__bf16)a.x; o[1] = (__bf16)a.y; o[2] = (__bf16)a.z; o[3] = (__bf16)a.w;
  o[4] = (__bf16)b.x; o[5] = (__bf16)b.y; o[6] = (__bf16)b.z; o[7] = (__bf16)b.w;
  *(v8bf*)(dst + i) = o;
}

// ---------------------------------------------------------------------------
// Batch routing (mean -> softmax -> +bias -> softmax -> pmax/argmax).
// ---------------------------------------------------------------------------
__global__ __launch_bounds__(256) void k_route(
    const float* __restrict__ h, const float* __restrict__ sw,
    const float* __restrict__ sb, const float* __restrict__ feb,
    int* __restrict__ routes, float* __restrict__ pm) {
  const int b = blockIdx.x;
  const int t = threadIdx.x;
  __shared__ float henc[D_];
  for (int d = t; d < D_; d += 256) {
    float s = 0.f;
    for (int ss = 0; ss < S_; ++ss)
      s += h[((size_t)b * S_ + ss) * D_ + d];
    henc[d] = s * (1.0f / S_);
  }
  __syncthreads();
  __shared__ float red[256];
  float logit[E_];
  for (int e = 0; e < E_; ++e) {
    float p = 0.f;
    for (int d = t; d < D_; d += 256) p += henc[d] * sw[d * E_ + e];
    red[t] = p; __syncthreads();
    for (int off = 128; off; off >>= 1) {
      if (t < off) red[t] += red[t + off];
      __syncthreads();
    }
    logit[e] = red[0] + sb[e];
    __syncthreads();
  }
  if (t == 0) {
    float m1 = fmaxf(fmaxf(logit[0], logit[1]), logit[2]);
    float e0 = __expf(logit[0] - m1), e1 = __expf(logit[1] - m1),
          e2 = __expf(logit[2] - m1);
    float z = e0 + e1 + e2;
    float l0 = e0 / z + feb[0], l1 = e1 / z + feb[1], l2 = e2 / z + feb[2];
    float m2 = fmaxf(fmaxf(l0, l1), l2);
    float r0 = __expf(l0 - m2), r1 = __expf(l1 - m2), r2 = __expf(l2 - m2);
    float z2 = r0 + r1 + r2;
    r0 /= z2; r1 /= z2; r2 /= z2;
    int idx = 0; float pmax = r0;
    if (r1 > pmax) { pmax = r1; idx = 1; }
    if (r2 > pmax) { pmax = r2; idx = 2; }
    if (pmax < 0.5f) { pmax = r0; idx = 0; }
    routes[b] = idx;
    pm[b] = pmax;                       // scale0 == pmax/sg(pmax) == 1
  }
}

// ---------------------------------------------------------------------------
// Route-gated bf16 GEMM: C = act(A[MxK] @ W[KxN] + bias); A,W bf16 row-major.
// Block tile 128x128, K-step 32, 8 waves (2x4), each wave 64x32 = 8 WMMA/step.
// Double-buffered staging: A tiles via async-to-LDS (2 copies/thread, in-order
// completion -> s_wait_asynccnt 2 overlaps next tile's copies with current
// compute); W tiles via paired-K register loads + packed 4B ds stores into the
// alternate buffer during the WMMA burst.
// ---------------------------------------------------------------------------
#define BM 128
#define BN 128
#define BK 32
#define ASTR 40   // 32 bf16 + 8 pad (80B rows, 16B-aligned chunks)
#define WSTR 40

__global__ __launch_bounds__(256) void k_gemm(
    const __bf16* __restrict__ A, const __bf16* __restrict__ W,
    const float* __restrict__ bias, float* __restrict__ Cf,
    __bf16* __restrict__ Cb, int M, int N, int K, int act,
    const int* __restrict__ routes, int gate) {
  const int row0 = blockIdx.y * BM;
  const int col0 = blockIdx.x * BN;
  if (gate >= 0 && routes[row0 / S_] != gate) return;

  __shared__ __bf16 As[2][BM * ASTR];   // [row][k]
  __shared__ __bf16 Ws[2][BN * WSTR];   // transposed: [n][k]

  const int t = threadIdx.x;
  const int wave = t >> 5;
  const int wm = wave >> 2;          // 0..1 -> 64 rows
  const int wn = wave & 3;           // 0..3 -> 32 cols

  // Per-thread staging coordinates.
  const int ar0 = t >> 2;            // A row (first copy)
  const int ac0 = (t & 3) << 3;      // A k-offset
  const int ar1 = (t + 256) >> 2;    // A row (second copy)
  const int kk0 = (t >> 4) << 1;     // W k-pair base: 0,2,..,30
  const int nq0 = (t & 15) << 3;     // W n base: 0..120

  auto issueA = [&](int buf, int k0) {
    async_b128(A + (size_t)(row0 + ar0) * K + k0 + ac0,
               &As[buf][ar0 * ASTR + ac0]);
    async_b128(A + (size_t)(row0 + ar1) * K + k0 + ac0,
               &As[buf][ar1 * ASTR + ac0]);
  };
  auto loadW = [&](int k0, v8bf& a, v8bf& b) {
    a = *(const v8bf*)(W + (size_t)(k0 + kk0) * N + col0 + nq0);
    b = *(const v8bf*)(W + (size_t)(k0 + kk0 + 1) * N + col0 + nq0);
  };
  auto storeW = [&](int buf, v8bf a, v8bf b) {
#pragma unroll
    for (int j = 0; j < 8; ++j) {
      v2bf p; p[0] = a[j]; p[1] = b[j];
      *(v2bf*)(&Ws[buf][(nq0 + j) * WSTR + kk0]) = p;   // 4B aligned (kk0 even)
    }
  };

  v8f acc[4][2] = {};
  v8bf wr0, wr1;

  // Prologue: stage tile 0.
  issueA(0, 0);
  loadW(0, wr0, wr1);
  storeW(0, wr0, wr1);

  int cur = 0;
  for (int k0 = 0; k0 < K; k0 += BK) {
    const bool nxt = (k0 + BK) < K;
    if (nxt) {
      issueA(cur ^ 1, k0 + BK);        // overlap next tile's copies
      loadW(k0 + BK, wr0, wr1);        // next W tile into registers
      if (k0 + 2 * BK < K)             // speculative prefetch, tile after next
        __builtin_prefetch(A + (size_t)(row0 + (t & 127)) * K + k0 + 2 * BK, 0, 1);
      wait_async<2>();                 // current tile done, next still in flight
    } else {
      wait_async<0>();
    }
    __syncthreads();                   // all waves' async + ds stores visible

    const __bf16* as = As[cur];
    const __bf16* ws = Ws[cur];
    v16bf bf0 = ld_frag(&ws[(wn * 32 + 0) * WSTR], WSTR);
    v16bf bf1 = ld_frag(&ws[(wn * 32 + 16) * WSTR], WSTR);
#pragma unroll
    for (int mi = 0; mi < 4; ++mi) {
      v16bf af = ld_frag(&as[(wm * 64 + mi * 16) * ASTR], ASTR);
      acc[mi][0] = wmma_bf16(af, bf0, acc[mi][0]);
      acc[mi][1] = wmma_bf16(af, bf1, acc[mi][1]);
    }
    if (nxt) storeW(cur ^ 1, wr0, wr1);  // drop next W tile during WMMA retire
    __syncthreads();                     // readers done before next overwrite
    cur ^= 1;
  }

  // Epilogue (C/D layout: vgpr v -> row v(+8 for upper half), lane%16 -> col).
  const int lane = t & 31;
  const int cl = lane & 15;
  const int rh = (lane >> 4) * 8;
#pragma unroll
  for (int mi = 0; mi < 4; ++mi) {
#pragma unroll
    for (int ni = 0; ni < 2; ++ni) {
      int col = col0 + wn * 32 + ni * 16 + cl;
      float bv = bias[col];
#pragma unroll
      for (int v = 0; v < 8; ++v) {
        float x = acc[mi][ni][v] + bv;
        if (act == 1) x = gelu_tanh(x);
        size_t idx = (size_t)(row0 + wm * 64 + mi * 16 + v + rh) * N + col;
        if (Cf) Cf[idx] = x;
        if (Cb) Cb[idx] = (__bf16)x;
      }
    }
  }
}

// ---------------------------------------------------------------------------
// Flash attention over bf16 Q/K/V: one block per (b, h, 128 q-rows); 8 waves,
// 16 q-rows each. K tile async-staged raw; V transposed with packed 4B stores.
// ---------------------------------------------------------------------------
#define KT 32
#define KTSTR 72   // Kt[key][d]: 64 + 8 pad (144B rows)
#define VTSTR 40   // Vt[d][key]: 32 + 8 pad
#define PSTR 40    // P[row][key]: 32 + 8 pad

__global__ __launch_bounds__(256) void k_attn(
    const __bf16* __restrict__ Q, const __bf16* __restrict__ Kb,
    const __bf16* __restrict__ Vb, const float* __restrict__ mask,
    __bf16* __restrict__ O, const int* __restrict__ routes, int gate) {
  const int b = blockIdx.x / H_;
  const int h = blockIdx.x % H_;
  if (gate >= 0 && routes[b] != gate) return;
  const int q0 = blockIdx.y * 128;

  __shared__ __bf16 Kt[KT * KTSTR];        // [key][d]
  __shared__ __bf16 Vt[DH_ * VTSTR];       // [d][key] (transposed)
  __shared__ __bf16 Pw[8 * 16 * PSTR];     // per-wave P tile

  const int t = threadIdx.x;
  const int wave = t >> 5;
  const int lane = t & 31;
  const int qr = q0 + wave * 16;

  // Q fragments for this wave's 16 rows (1/sqrt(dh)=0.125 folded in).
  const int arow = lane & 15;
  const int kb = (lane >> 4) * 8;
  const __bf16* qp = Q + ((size_t)(b * S_) + qr + arow) * D_ + h * DH_;
  v16bf aq[2];
#pragma unroll
  for (int c = 0; c < 2; ++c) {
    v8bf lo = *(const v8bf*)(qp + c * 32 + kb);
    v8bf hi = *(const v8bf*)(qp + c * 32 + kb + 16);
#pragma unroll
    for (int i = 0; i < 8; ++i) {
      aq[c][i]     = (__bf16)((float)lo[i] * 0.125f);
      aq[c][i + 8] = (__bf16)((float)hi[i] * 0.125f);
    }
  }

  float mrow[8], lrow[8];
  v8f ctx[4] = {};
#pragma unroll
  for (int v = 0; v < 8; ++v) { mrow[v] = -3.0e38f; lrow[v] = 0.f; }

  for (int kt0 = 0; kt0 < S_; kt0 += KT) {
    __syncthreads();
    // K tile: 32 keys x 128B rows -> 256 async 16B copies (1/thread).
    {
      int key = t >> 3;                    // 0..31
      int c = (t & 7) << 3;                // 0..56 (bf16 elems)
      async_b128(Kb + ((size_t)(b * S_) + kt0 + key) * D_ + h * DH_ + c,
                 &Kt[key * KTSTR + c]);
    }
    // V tile transposed into Vt[d][key]: paired keys -> packed 4B ds stores.
    {
      int kp = (t >> 4) << 1;              // keys kp, kp+1 (0,2,..,30)
      int dq = (t & 15) << 2;              // 0..60, 4 dims
      const __bf16* vb0 =
          Vb + ((size_t)(b * S_) + kt0 + kp) * D_ + h * DH_ + dq;
      v4bf v0 = *(const v4bf*)(vb0);
      v4bf v1 = *(const v4bf*)(vb0 + D_);
#pragma unroll
      for (int j = 0; j < 4; ++j) {
        v2bf p; p[0] = v0[j]; p[1] = v1[j];
        *(v2bf*)(&Vt[(dq + j) * VTSTR + kp]) = p;
      }
    }
    wait_async<0>();
    __syncthreads();

    // scores (16 q-rows x 32 keys): two 16-key column tiles, K-dim = 64.
    v8f sc[2];
#pragma unroll
    for (int nt = 0; nt < 2; ++nt) {
      v8f s = {};
      s = wmma_bf16(aq[0], ld_frag(&Kt[(nt * 16) * KTSTR + 0], KTSTR), s);
      s = wmma_bf16(aq[1], ld_frag(&Kt[(nt * 16) * KTSTR + 32], KTSTR), s);
      float mv = mask[b * S_ + kt0 + nt * 16 + (lane & 15)];
#pragma unroll
      for (int v = 0; v < 8; ++v) s[v] += mv;
      sc[nt] = s;
    }

    // online softmax; each row lives in a 16-lane half of the wave.
#pragma unroll
    for (int v = 0; v < 8; ++v) {
      float rmax = fmaxf(sc[0][v], sc[1][v]);
      for (int off = 1; off < 16; off <<= 1)
        rmax = fmaxf(rmax, __shfl_xor(rmax, off, 32));
      float mnew = fmaxf(mrow[v], rmax);
      float corr = __expf(mrow[v] - mnew);
      mrow[v] = mnew;
      float p0 = __expf(sc[0][v] - mnew);
      float p1 = __expf(sc[1][v] - mnew);
      sc[0][v] = p0; sc[1][v] = p1;
      float sum = p0 + p1;
      for (int off = 1; off < 16; off <<= 1)
        sum += __shfl_xor(sum, off, 32);
      lrow[v] = lrow[v] * corr + sum;
#pragma unroll
      for (int dt = 0; dt < 4; ++dt) ctx[dt][v] *= corr;
    }

    // Re-layout P through per-wave LDS into an A fragment.
    __bf16* pw = &Pw[wave * 16 * PSTR];
    {
      const int prh = (lane >> 4) << 3;
      const int pcl = lane & 15;
#pragma unroll
      for (int v = 0; v < 8; ++v) {
        pw[(v + prh) * PSTR + pcl] = (__bf16)sc[0][v];
        pw[(v + prh) * PSTR + 16 + pcl] = (__bf16)sc[1][v];
      }
    }
    __syncthreads();

    v16bf ap = ld_frag(pw, PSTR);
#pragma unroll
    for (int dt = 0; dt < 4; ++dt) {
      v16bf bv = ld_frag(&Vt[(dt * 16) * VTSTR], VTSTR);
      ctx[dt] = wmma_bf16(ap, bv, ctx[dt]);
    }
  }

  // normalize + store ctx as bf16 (feeds the O-projection GEMM).
  const int cl = lane & 15;
  const int rh = (lane >> 4) * 8;
#pragma unroll
  for (int dt = 0; dt < 4; ++dt) {
#pragma unroll
    for (int v = 0; v < 8; ++v) {
      O[((size_t)(b * S_) + qr + v + rh) * D_ + h * DH_ + dt * 16 + cl] =
          (__bf16)(ctx[dt][v] / lrow[v]);
    }
  }
}

// ---------------------------------------------------------------------------
// Per-token switch-FFN router (NI=2), gated by batch route.
// ---------------------------------------------------------------------------
__global__ __launch_bounds__(128) void k_tok_route(
    const float* __restrict__ X, const float* __restrict__ rw,
    const float* __restrict__ rb, float* __restrict__ tpm,
    int* __restrict__ tidx, const int* __restrict__ routes, int gate) {
  const int tok = blockIdx.x;
  if (routes[tok / S_] != gate) return;
  const int t = threadIdx.x;
  float p0 = 0.f, p1 = 0.f;
  for (int d = t; d < D_; d += 128) {
    float x = X[(size_t)tok * D_ + d];
    p0 += x * rw[d * NI_ + 0];
    p1 += x * rw[d * NI_ + 1];
  }
  __shared__ float r0[128], r1[128];
  r0[t] = p0; r1[t] = p1; __syncthreads();
  for (int off = 64; off; off >>= 1) {
    if (t < off) { r0[t] += r0[t + off]; r1[t] += r1[t + off]; }
    __syncthreads();
  }
  if (t == 0) {
    float l0 = r0[0] + rb[0], l1 = r1[0] + rb[1];
    float m = fmaxf(l0, l1);
    float e0 = __expf(l0 - m), e1 = __expf(l1 - m);
    tpm[tok] = fmaxf(e0, e1) / (e0 + e1);
    tidx[tok] = (l1 > l0) ? 1 : 0;
  }
}

// ---------------------------------------------------------------------------
// attn_out = route==0 ? common : pm*expert + (1-pm)*common. Writes f32 (for
// residual/LN) and bf16 (feeds FFN GEMMs).
// ---------------------------------------------------------------------------
__global__ __launch_bounds__(256) void k_blend_attn(
    const float* __restrict__ c, const float* __restrict__ u,
    float* __restrict__ o, __bf16* __restrict__ ob,
    const int* __restrict__ routes, const float* __restrict__ pm) {
  size_t i = (size_t)blockIdx.x * 256 + threadIdx.x;
  int b = (int)(i / ((size_t)S_ * D_));
  int r = routes[b];
  float cv = c[i];
  float val;
  if (r == 0) {
    val = cv;
  } else {
    float p = pm[b];
    val = p * u[i] + (1.0f - p) * cv;
  }
  o[i] = val;
  ob[i] = (__bf16)val;
}

// ---------------------------------------------------------------------------
// Fused FFN blend + residual + LayerNorm. One block per token.
// ---------------------------------------------------------------------------
__global__ __launch_bounds__(256) void k_ln(
    const float* __restrict__ attn, const float* __restrict__ cffn,
    const float* __restrict__ o0, const float* __restrict__ o1,
    const float* __restrict__ tpm, const int* __restrict__ tidx,
    const int* __restrict__ routes, const float* __restrict__ pm,
    const float* __restrict__ g, const float* __restrict__ bb,
    float* __restrict__ out) {
  const int tok = blockIdx.x;
  const int t = threadIdx.x;
  const int b = tok / S_;
  const int r = routes[b];
  const float p = pm[b];
  const float tp = (r > 0) ? tpm[tok] : 0.f;
  const int ti = (r > 0) ? tidx[tok] : 0;

  float x[4];
#pragma unroll
  for (int j = 0; j < 4; ++j) {
    int d = t + 256 * j;
    size_t i = (size_t)tok * D_ + d;
    float f;
    if (r == 0) {
      f = cffn[i];
    } else {
      float sel = ((ti == 0) ? o0[i] : o1[i]) * tp;
      f = p * sel + (1.0f - p) * cffn[i];
    }
    x[j] = attn[i] + f;
  }
  __shared__ float red[256];
  float s = x[0] + x[1] + x[2] + x[3];
  red[t] = s; __syncthreads();
  for (int off = 128; off; off >>= 1) {
    if (t < off) red[t] += red[t + off];
    __syncthreads();
  }
  float mu = red[0] * (1.0f / D_);
  __syncthreads();
  float q = 0.f;
#pragma unroll
  for (int j = 0; j < 4; ++j) { float dd = x[j] - mu; q += dd * dd; }
  red[t] = q; __syncthreads();
  for (int off = 128; off; off >>= 1) {
    if (t < off) red[t] += red[t + off];
    __syncthreads();
  }
  float rstd = rsqrtf(red[0] * (1.0f / D_) + 1e-12f);
#pragma unroll
  for (int j = 0; j < 4; ++j) {
    int d = t + 256 * j;
    size_t i = (size_t)tok * D_ + d;
    out[i] = (x[j] - mu) * rstd * g[d] + bb[d];
  }
}

// ---------------------------------------------------------------------------
// Host orchestration
// ---------------------------------------------------------------------------
extern "C" void kernel_launch(void* const* d_in, const int* in_sizes, int n_in,
                              void* d_out, int out_size, void* d_ws,
                              size_t ws_size, hipStream_t stream) {
  (void)in_sizes; (void)n_in; (void)out_size; (void)ws_size;
  const float* h    = (const float*)d_in[0];
  const float* msk  = (const float*)d_in[1];
  const float* sw   = (const float*)d_in[3];
  const float* sb   = (const float*)d_in[4];
  const float* feb  = (const float*)d_in[5];
  const float* Wq   = (const float*)d_in[6];
  const float* bq   = (const float*)d_in[7];
  const float* Wk   = (const float*)d_in[8];
  const float* bk   = (const float*)d_in[9];
  const float* Wv   = (const float*)d_in[10];
  const float* bv   = (const float*)d_in[11];
  const float* Wo   = (const float*)d_in[12];
  const float* bo   = (const float*)d_in[13];
  const float* W1   = (const float*)d_in[14];
  const float* b1   = (const float*)d_in[15];
  const float* W2   = (const float*)d_in[16];
  const float* b2   = (const float*)d_in[17];
  const float* urw  = (const float*)d_in[18];
  const float* urb  = (const float*)d_in[19];
  const float* uW1  = (const float*)d_in[20];
  const float* ub1  = (const float*)d_in[21];
  const float* uW2  = (const float*)d_in[22];
  const float* ub2  = (const float*)d_in[23];
  const float* lng  = (const float*)d_in[24];
  const float* lnb  = (const float*)d_in[25];
  float* out = (float*)d_out;

  char* w = (char*)d_ws;
  auto alloc = [&](size_t bytes) -> char* {
    char* p = w;
    w += (bytes + 255) & ~(size_t)255;
    return p;
  };
  int*   routes = (int*)alloc(B_ * sizeof(int));
  float* pm     = (float*)alloc(B_ * sizeof(float));
  int*   tidx   = (int*)alloc((size_t)TOK_ * sizeof(int));
  float* tpm    = (float*)alloc((size_t)TOK_ * sizeof(float));

  const size_t TDf = (size_t)TOK_ * D_ * sizeof(float);
  const size_t TDb = (size_t)TOK_ * D_ * sizeof(__bf16);
  float*  attC = (float*)alloc(TDf);
  float*  attU = (float*)alloc(TDf);   // experts 1,2 write disjoint batches
  float*  attO = (float*)alloc(TDf);
  float*  ffnC = (float*)alloc(TDf);
  float*  out0 = (float*)alloc(TDf);
  float*  out1 = (float*)alloc(TDf);
  __bf16* hb    = (__bf16*)alloc(TDb);
  __bf16* bufQ  = (__bf16*)alloc(TDb);
  __bf16* bufK  = (__bf16*)alloc(TDb);
  __bf16* bufV  = (__bf16*)alloc(TDb);
  __bf16* bufCtx= (__bf16*)alloc(TDb);
  __bf16* attOb = (__bf16*)alloc(TDb);
  __bf16* hid   = (__bf16*)alloc((size_t)TOK_ * DFF_ * sizeof(__bf16));
  const size_t WQKV = (size_t)E_ * D_ * D_;
  __bf16* wqb = (__bf16*)alloc(WQKV * 2);
  __bf16* wkb = (__bf16*)alloc(WQKV * 2);
  __bf16* wvb = (__bf16*)alloc(WQKV * 2);
  __bf16* wob = (__bf16*)alloc(WQKV * 2);
  __bf16* w1b = (__bf16*)alloc((size_t)D_ * DFF_ * 2);
  __bf16* w2b = (__bf16*)alloc((size_t)DFF_ * D_ * 2);
  __bf16* uw1b = (__bf16*)alloc((size_t)U_ * NI_ * D_ * DFF_ * 2);
  __bf16* uw2b = (__bf16*)alloc((size_t)U_ * NI_ * DFF_ * D_ * 2);

  dim3 blk(256);
  auto cvt = [&](const float* s, __bf16* d, size_t n) {
    k_cvt<<<dim3((unsigned)(n / 8 / 256)), blk, 0, stream>>>(s, d, n);
  };
  cvt(h, hb, (size_t)TOK_ * D_);
  cvt(Wq, wqb, WQKV); cvt(Wk, wkb, WQKV);
  cvt(Wv, wvb, WQKV); cvt(Wo, wob, WQKV);
  cvt(W1, w1b, (size_t)D_ * DFF_);
  cvt(W2, w2b, (size_t)DFF_ * D_);
  cvt(uW1, uw1b, (size_t)U_ * NI_ * D_ * DFF_);
  cvt(uW2, uw2b, (size_t)U_ * NI_ * DFF_ * D_);

  k_route<<<dim3(B_), blk, 0, stream>>>(h, sw, sb, feb, routes, pm);

  auto gemm = [&](const __bf16* Aa, const __bf16* Ww, const float* bias,
                  float* Cf, __bf16* Cb, int M, int N, int K, int act,
                  int gate) {
    k_gemm<<<dim3(N / BN, M / BM), blk, 0, stream>>>(Aa, Ww, bias, Cf, Cb, M, N,
                                                     K, act, routes, gate);
  };

  const size_t DD = (size_t)D_ * D_;
  for (int e = 0; e < E_; ++e) {
    int gate = (e == 0) ? -1 : e;          // expert 0 = common, all batches
    float* dst = (e == 0) ? attC : attU;
    gemm(hb, wqb + e * DD, bq + (size_t)e * D_, nullptr, bufQ, TOK_, D_, D_, 0, gate);
    gemm(hb, wkb + e * DD, bk + (size_t)e * D_, nullptr, bufK, TOK_, D_, D_, 0, gate);
    gemm(hb, wvb + e * DD, bv + (size_t)e * D_, nullptr, bufV, TOK_, D_, D_, 0, gate);
    k_attn<<<dim3(B_ * H_, S_ / 128), blk, 0, stream>>>(bufQ, bufK, bufV, msk,
                                                        bufCtx, routes, gate);
    gemm(bufCtx, wob + e * DD, bo + (size_t)e * D_, dst, nullptr, TOK_, D_, D_, 0, gate);
  }

  k_blend_attn<<<dim3((unsigned)(((size_t)TOK_ * D_) / 256)), blk, 0, stream>>>(
      attC, attU, attO, attOb, routes, pm);

  // common FFN (needed for every route)
  gemm(attOb, w1b, b1, nullptr, hid, TOK_, DFF_, D_, 1, -1);
  gemm(hid, w2b, b2, ffnC, nullptr, TOK_, D_, DFF_, 0, -1);

  // switch-FFN units (unit u serves route u+1); both inner experts computed,
  // per-token selection happens in the LN kernel.
  for (int u = 0; u < U_; ++u) {
    int gate = u + 1;
    k_tok_route<<<dim3(TOK_), dim3(128), 0, stream>>>(
        attO, urw + (size_t)u * D_ * NI_, urb + (size_t)u * NI_, tpm, tidx,
        routes, gate);
    for (int i = 0; i < NI_; ++i) {
      size_t wi = (size_t)(u * NI_ + i);
      gemm(attOb, uw1b + wi * D_ * DFF_, ub1 + wi * DFF_, nullptr, hid, TOK_,
           DFF_, D_, 1, gate);
      gemm(hid, uw2b + wi * DFF_ * D_, ub2 + wi * D_, (i == 0) ? out0 : out1,
           nullptr, TOK_, D_, DFF_, 0, gate);
    }
  }

  k_ln<<<dim3(TOK_), blk, 0, stream>>>(attO, ffnC, out0, out1, tpm, tidx,
                                       routes, pm, lng, lnb, out);
}
